// NSAttention_27041114095963
// MI455X (gfx1250) — matrix operand order repaired and checked by reading, not compiled, not run on hardware
//
#include <hip/hip_runtime.h>

#define S_    2048
#define DM_   1024
#define H_    16
#define HD_   64
#define BS_   32
#define NB_   64
#define TOPK_ 8
#define WIN_  256
#define SCALE_ 0.125f
#define NEGV  -1e30f

typedef __attribute__((ext_vector_type(16))) __bf16 v16bf;
typedef __attribute__((ext_vector_type(8)))  float  v8f;

union Frag { v16bf h; unsigned int u[8]; };

__device__ __forceinline__ unsigned short f2bf(float f) {
  unsigned u = __float_as_uint(f);
  return (unsigned short)((u + 0x7FFFu + ((u >> 16) & 1u)) >> 16);
}

// A-fragment (16x32 bf16): lane = m(0..15) + 16*half; VGPR v holds K pair at
// k = (v>=4)*16 + half*8 + (v&3)*2   (ISA 7.12.2, 16-bit A 16x32 table)
__device__ __forceinline__ void load_fragA(const unsigned short* p, int ld, Frag& f) {
  int lane = threadIdx.x & 31;
  const unsigned short* row = p + (size_t)(lane & 15) * ld;
  int half = lane >> 4;
#pragma unroll
  for (int v = 0; v < 8; ++v) {
    int k = ((v & 4) << 2) + (half << 3) + ((v & 3) << 1);
    f.u[v] = *(const unsigned int*)(row + k);
  }
}

// B-fragment (32x16 bf16) from N-major ("Wt"-style [N][K]) storage:
// lane = n(0..15) + 16*half; VGPR v holds K pair at k = (v>=4)*16 + half*8 + (v&3)*2
// (ISA 7.12.4/7.12.5 B-matrix striping)
__device__ __forceinline__ void load_fragB(const unsigned short* p, int ld, Frag& f) {
  int lane = threadIdx.x & 31;
  const unsigned short* row = p + (size_t)(lane & 15) * ld;
  int half = lane >> 4;
#pragma unroll
  for (int v = 0; v < 8; ++v) {
    int k = ((v & 4) << 2) + (half << 3) + ((v & 3) << 1);
    f.u[v] = *(const unsigned int*)(row + k);
  }
}

__device__ __forceinline__ v8f wmma_bf16(const Frag& a, const Frag& b, v8f c) {
  return __builtin_amdgcn_wmma_f32_16x16x32_bf16(false, a.h, false, b.h,
                                                 (short)0, c, false, false);
}

// ---------------- generic WMMA GEMM: C = act(A @ Bt^T * alpha + bias) --------
// A [M,K] bf16 row-major (lda), Bt [N,K] bf16 row-major (ldb), C f32 or bf16.
// 128 threads = 4 waves stacked along M; each wave computes a 64x32 tile
// (4x2 WMMA subtiles, A-fragments streamed so only one is live at a time:
//  peak ~150 VGPRs -> no spills, ~22 flop/byte).
// Block tile = 256(M) x 32(N). blockIdx.z batches heads via *Head strides.
__global__ void __launch_bounds__(128, 1)
gemm_wmma(const unsigned short* __restrict__ A, long aHead, int lda,
          const unsigned short* __restrict__ Bt, long bHead, int ldb,
          const float* __restrict__ bias,
          void* __restrict__ Cout, long cHead, int ldc,
          int K, float alpha, int act, int outBf16) {
  int wave = threadIdx.x >> 5;
  int lane = threadIdx.x & 31;
  int nBase = blockIdx.x * 32;
  int mBase = blockIdx.y * 256 + wave * 64;
  long hz = blockIdx.z;
  A += hz * aHead;
  Bt += hz * bHead;
  v8f acc[4][2] = {};
  for (int k0 = 0; k0 < K; k0 += 32) {
    Frag b0, b1;
    load_fragB(Bt + (size_t)nBase * ldb + k0, ldb, b0);
    load_fragB(Bt + (size_t)(nBase + 16) * ldb + k0, ldb, b1);
#pragma unroll
    for (int i = 0; i < 4; ++i) {
      Frag a;
      load_fragA(A + (size_t)(mBase + i * 16) * lda + k0, lda, a);
      acc[i][0] = wmma_bf16(a, b0, acc[i][0]);
      acc[i][1] = wmma_bf16(a, b1, acc[i][1]);
    }
  }
  int half = lane >> 4, col = lane & 15;
#pragma unroll
  for (int i = 0; i < 4; ++i)
#pragma unroll
    for (int j = 0; j < 2; ++j) {
      int n = nBase + j * 16 + col;
      float bv = bias ? bias[n] : 0.f;
#pragma unroll
      for (int v = 0; v < 8; ++v) {
        int m = mBase + i * 16 + v + 8 * half;
        float f = acc[i][j][v] * alpha + bv;
        if (act == 1) f = 0.5f * f * (1.f + erff(f * 0.70710678118f));
        size_t idx = (size_t)hz * cHead + (size_t)m * ldc + n;
        if (outBf16) ((unsigned short*)Cout)[idx] = f2bf(f);
        else         ((float*)Cout)[idx] = f;
      }
    }
}

// ---------------- flash attention, one wave per 16-query tile ----------------
// mode 0: per-row block-selection mask (64-bit per row), chunk-skip on union
// mode 1: causal sliding window (WIN_)
// mode 2: dense small (nKeys keys), used for the compressed branch
__global__ void flash_wmma(const unsigned short* __restrict__ qb, int ldq,
                           const unsigned short* __restrict__ kp, long kHead, int ldk,
                           const unsigned short* __restrict__ vt, long vHead, int ldvt,
                           float* __restrict__ outp, int ldo,
                           const unsigned long long* __restrict__ msk,
                           int nKeys, int nQ, int mode, float scale) {
  __shared__ __align__(16) unsigned short plds[4][16 * 32];
  int wave = threadIdx.x >> 5, lane = threadIdx.x & 31;
  int h = blockIdx.y;
  int sBase = (blockIdx.x * 4 + wave) * 16;
  int half = lane >> 4, col = lane & 15;

  const unsigned short* qh = qb + (size_t)h * HD_;
  const unsigned short* kh = kp + (size_t)h * kHead;
  const unsigned short* vh = vt + (size_t)h * vHead;

  Frag q0, q1;
  load_fragA(qh + (size_t)sBase * ldq + 0,  ldq, q0);
  load_fragA(qh + (size_t)sBase * ldq + 32, ldq, q1);

  v8f o0 = {}, o1 = {}, o2 = {}, o3 = {};
  float mrow[8], lsum[8];
  unsigned long long rm[8];
#pragma unroll
  for (int v = 0; v < 8; ++v) { mrow[v] = -3.0e38f; lsum[v] = 0.f; rm[v] = 0ull; }

  unsigned long long uni = ~0ull;
  if (mode == 0) {
#pragma unroll
    for (int v = 0; v < 8; ++v)
      rm[v] = msk[(size_t)h * nQ + sBase + v + 8 * half];
    unsigned long long mine = (lane < 16) ? msk[(size_t)h * nQ + sBase + lane] : 0ull;
    unsigned int lo = (unsigned int)mine, hi = (unsigned int)(mine >> 32);
#pragma unroll
    for (int off = 16; off >= 1; off >>= 1) { lo |= __shfl_xor(lo, off); hi |= __shfl_xor(hi, off); }
    uni = ((unsigned long long)hi << 32) | lo;
  }

  int tLo = 0, tHi = nKeys - 32;
  if (mode == 1) {
    int l = sBase - (WIN_ - 1); if (l < 0) l = 0;
    tLo = l & ~31;
    tHi = ((sBase + 15) >> 5) << 5;
  }

  for (int t0 = tLo; t0 <= tHi; t0 += 32) {
    if (mode == 0 && !((uni >> (t0 >> 5)) & 1ull)) continue;

    Frag b00, b01, b10, b11;
    load_fragB(kh + (size_t)t0 * ldk + 0,  ldk, b00);
    load_fragB(kh + (size_t)t0 * ldk + 32, ldk, b01);
    load_fragB(kh + (size_t)(t0 + 16) * ldk + 0,  ldk, b10);
    load_fragB(kh + (size_t)(t0 + 16) * ldk + 32, ldk, b11);

    v8f s0 = {}, s1 = {};
    s0 = wmma_bf16(q0, b00, s0); s0 = wmma_bf16(q1, b01, s0);
    s1 = wmma_bf16(q0, b10, s1); s1 = wmma_bf16(q1, b11, s1);

#pragma unroll
    for (int v = 0; v < 8; ++v) {
      int row = v + 8 * half;
      int s = sBase + row;
      float x0 = s0[v] * scale, x1 = s1[v] * scale;
      if (mode == 0) {
        if (!((rm[v] >> (t0 >> 5)) & 1ull)) { x0 = NEGV; x1 = NEGV; }
      } else if (mode == 1) {
        int ta = t0 + col, tb = ta + 16;
        if (!(ta <= s && s - ta < WIN_)) x0 = NEGV;
        if (!(tb <= s && s - tb < WIN_)) x1 = NEGV;
      }
      float tm = fmaxf(x0, x1);
#pragma unroll
      for (int off = 8; off >= 1; off >>= 1) tm = fmaxf(tm, __shfl_xor(tm, off));
      float nm = fmaxf(mrow[v], tm);
      float cf = __expf(mrow[v] - nm);
      mrow[v] = nm;
      float p0 = __expf(x0 - nm), p1 = __expf(x1 - nm);
      float ps = p0 + p1;
#pragma unroll
      for (int off = 8; off >= 1; off >>= 1) ps += __shfl_xor(ps, off);
      lsum[v] = lsum[v] * cf + ps;
      o0[v] *= cf; o1[v] *= cf; o2[v] *= cf; o3[v] *= cf;
      plds[wave][row * 32 + col]      = f2bf(p0);
      plds[wave][row * 32 + col + 16] = f2bf(p1);
    }
    asm volatile("s_wait_dscnt 0x0" ::: "memory");  // wave-local LDS RAW fence
    Frag pf;
    load_fragA(&plds[wave][0], 32, pf);
    Frag vf0, vf1, vf2, vf3;
    load_fragB(vh + (size_t)0  * ldvt + t0, ldvt, vf0);
    load_fragB(vh + (size_t)16 * ldvt + t0, ldvt, vf1);
    load_fragB(vh + (size_t)32 * ldvt + t0, ldvt, vf2);
    load_fragB(vh + (size_t)48 * ldvt + t0, ldvt, vf3);
    o0 = wmma_bf16(pf, vf0, o0);
    o1 = wmma_bf16(pf, vf1, o1);
    o2 = wmma_bf16(pf, vf2, o2);
    o3 = wmma_bf16(pf, vf3, o3);
  }

  float* oh = outp + (size_t)h * HD_;
#pragma unroll
  for (int v = 0; v < 8; ++v) {
    int row = sBase + v + 8 * half;
    float inv = 1.f / lsum[v];
    oh[(size_t)row * ldo + 0 * 16 + col] = o0[v] * inv;
    oh[(size_t)row * ldo + 1 * 16 + col] = o1[v] * inv;
    oh[(size_t)row * ldo + 2 * 16 + col] = o2[v] * inv;
    oh[(size_t)row * ldo + 3 * 16 + col] = o3[v] * inv;
  }
}

// ---------------- small utility kernels ----------------
__global__ void k_cvt(const float* __restrict__ in, unsigned short* __restrict__ out, int n) {
  int i = blockIdx.x * blockDim.x + threadIdx.x;
  if (i < n) out[i] = f2bf(in[i]);
}

// out[((b*OC+co)*OR + ro)] = bf16(in[b*ibs + ro*irs + co*ics])
__global__ void k_tconv(const float* __restrict__ in, unsigned short* __restrict__ out,
                        int total, int OR_, int OC_, long ibs, long irs, long ics) {
  int i = blockIdx.x * blockDim.x + threadIdx.x;
  if (i >= total) return;
  int ro = i % OR_;
  int t = i / OR_;
  int co = t % OC_;
  int b = t / OC_;
  out[i] = f2bf(in[(size_t)b * ibs + (size_t)ro * irs + (size_t)co * ics]);
}

// KB[h*NB+nb][t*HD+d] = bf16(k[(nb*BS+t)*DM + h*HD + d] + pos[t*HD+d])
__global__ void k_buildKB(const float* __restrict__ kf, const float* __restrict__ pos,
                          unsigned short* __restrict__ KB, int total) {
  int i = blockIdx.x * blockDim.x + threadIdx.x;
  if (i >= total) return;
  int cc = i & (BS_ * HD_ - 1);
  int rb = i >> 11;
  int d = cc & 63, t = cc >> 6;
  int nb = rb & 63, h = rb >> 6;
  KB[i] = f2bf(kf[(size_t)(nb * BS_ + t) * DM_ + h * HD_ + d] + pos[t * HD_ + d]);
}

// per-block key mean -> bf16 [H][NB][HD]
__global__ void k_kbm(const float* __restrict__ kf, unsigned short* __restrict__ kbmb) {
  int i = blockIdx.x * blockDim.x + threadIdx.x;
  if (i >= H_ * NB_ * HD_) return;
  int d = i & 63, nb = (i >> 6) & 63, h = i >> 12;
  float s = 0.f;
  for (int t = 0; t < BS_; ++t) s += kf[(size_t)(nb * BS_ + t) * DM_ + h * HD_ + d];
  kbmb[i] = f2bf(s * (1.f / BS_));
}

// top-8 of 64 block scores -> 64-bit mask; '>' keeps earliest index on ties (jax top_k)
__global__ void k_topk(const float* __restrict__ bs, unsigned long long* __restrict__ m, int total) {
  int i = blockIdx.x * blockDim.x + threadIdx.x;
  if (i >= total) return;
  const float* p = bs + (size_t)i * NB_;
  unsigned long long sel = 0ull;
  for (int t = 0; t < TOPK_; ++t) {
    float best = -3.4e38f; int bi = 0;
    for (int j = 0; j < NB_; ++j) {
      float vv = p[j];
      if (!((sel >> j) & 1ull) && vv > best) { best = vv; bi = j; }
    }
    sel |= (1ull << bi);
  }
  m[i] = sel;
}

__global__ void k_gate(const float* __restrict__ h1, const float* __restrict__ W2,
                       const float* __restrict__ b2, float* __restrict__ g) {
  int s = blockIdx.x; int lane = threadIdx.x;
#pragma unroll
  for (int o = 0; o < 3; ++o) {
    float acc = 0.f;
    for (int j = lane; j < DM_ / 2; j += 32) acc += h1[(size_t)s * (DM_ / 2) + j] * W2[j * 3 + o];
#pragma unroll
    for (int off = 16; off >= 1; off >>= 1) acc += __shfl_xor(acc, off);
    if (lane == 0) g[s * 3 + o] = 1.f / (1.f + __expf(-(acc + b2[o])));
  }
}

__global__ void k_fuse(const float* __restrict__ oc, const float* __restrict__ os,
                       const float* __restrict__ ow, const float* __restrict__ g,
                       unsigned short* __restrict__ out, int total) {
  int i = blockIdx.x * blockDim.x + threadIdx.x;
  if (i >= total) return;
  int s = i >> 10;
  float r = g[s * 3 + 0] * oc[i] + g[s * 3 + 1] * os[i] + g[s * 3 + 2] * ow[i];
  out[i] = f2bf(r);
}

// ---------------- host orchestration ----------------
extern "C" void kernel_launch(void* const* d_in, const int* in_sizes, int n_in,
                              void* d_out, int out_size, void* d_ws, size_t ws_size,
                              hipStream_t stream) {
  const float* x   = (const float*)d_in[0];
  const float* Wq  = (const float*)d_in[1];  const float* bq  = (const float*)d_in[2];
  const float* Wk  = (const float*)d_in[3];  const float* bk  = (const float*)d_in[4];
  const float* Wv  = (const float*)d_in[5];  const float* bv  = (const float*)d_in[6];
  const float* Wo  = (const float*)d_in[7];  const float* bo  = (const float*)d_in[8];
  const float* cW1 = (const float*)d_in[9];  const float* cb1 = (const float*)d_in[10];
  const float* cW2 = (const float*)d_in[11]; const float* cb2 = (const float*)d_in[12];
  const float* pos = (const float*)d_in[13];
  const float* gW1 = (const float*)d_in[14]; const float* gb1 = (const float*)d_in[15];
  const float* gW2 = (const float*)d_in[16]; const float* gb2 = (const float*)d_in[17];

  char* base = (char*)d_ws;
  size_t off = 0;
  auto alloc = [&](size_t bytes) -> char* {
    char* p = base + off;
    off += (bytes + 255) & ~(size_t)255;
    return p;
  };

  unsigned short* xb    = (unsigned short*)alloc((size_t)S_ * DM_ * 2);
  unsigned short* WqT   = (unsigned short*)alloc((size_t)DM_ * DM_ * 2);
  unsigned short* WkT   = (unsigned short*)alloc((size_t)DM_ * DM_ * 2);
  unsigned short* WvT   = (unsigned short*)alloc((size_t)DM_ * DM_ * 2);
  unsigned short* WoT   = (unsigned short*)alloc((size_t)DM_ * DM_ * 2);
  unsigned short* cW1T  = (unsigned short*)alloc((size_t)BS_ * HD_ * 4 * HD_ * 2);
  unsigned short* cW2T  = (unsigned short*)alloc((size_t)4 * HD_ * HD_ * 2);
  unsigned short* gW1T  = (unsigned short*)alloc((size_t)DM_ * (DM_ / 2) * 2);
  unsigned short* qb    = (unsigned short*)alloc((size_t)S_ * DM_ * 2);
  float*          kf    = (float*)alloc((size_t)S_ * DM_ * 4);
  float*          vf    = (float*)alloc((size_t)S_ * DM_ * 4);
  unsigned short* kb    = (unsigned short*)alloc((size_t)S_ * DM_ * 2);
  unsigned short* vtb   = (unsigned short*)alloc((size_t)H_ * HD_ * S_ * 2);
  unsigned short* KB    = (unsigned short*)alloc((size_t)H_ * NB_ * BS_ * HD_ * 2);
  float*          H1    = (float*)alloc((size_t)H_ * NB_ * 4 * HD_ * 4);
  unsigned short* H1b   = (unsigned short*)alloc((size_t)H_ * NB_ * 4 * HD_ * 2);
  float*          kcf   = (float*)alloc((size_t)H_ * NB_ * HD_ * 4);
  float*          vcf   = (float*)alloc((size_t)H_ * NB_ * HD_ * 4);
  unsigned short* kcb   = (unsigned short*)alloc((size_t)H_ * NB_ * HD_ * 2);
  unsigned short* vctb  = (unsigned short*)alloc((size_t)H_ * HD_ * NB_ * 2);
  unsigned short* kbmb  = (unsigned short*)alloc((size_t)H_ * NB_ * HD_ * 2);
  float*          bsc   = (float*)alloc((size_t)H_ * S_ * NB_ * 4);
  unsigned long long* msk = (unsigned long long*)alloc((size_t)H_ * S_ * 8);
  float*          outc  = (float*)alloc((size_t)S_ * DM_ * 4);
  float*          outs  = (float*)alloc((size_t)S_ * DM_ * 4);
  float*          outw  = (float*)alloc((size_t)S_ * DM_ * 4);
  float*          h1g   = (float*)alloc((size_t)S_ * (DM_ / 2) * 4);
  float*          gg    = (float*)alloc((size_t)S_ * 3 * 4);
  unsigned short* fusedb = (unsigned short*)alloc((size_t)S_ * DM_ * 2);

  auto gemm = [&](const unsigned short* A, long aH, int lda,
                  const unsigned short* Bt, long bH, int ldb,
                  const float* bias, void* C, long cH, int ldc,
                  int M, int N, int K, float alpha, int act, int obf, int Z) {
    dim3 g(N / 32, M / 256, Z);
    gemm_wmma<<<g, 128, 0, stream>>>(A, aH, lda, Bt, bH, ldb, bias, C, cH, ldc,
                                     K, alpha, act, obf);
  };
  auto grid1 = [](int n) { return dim3((n + 255) / 256); };

  // ---- convert x and pre-transpose weights to bf16 ----
  k_cvt<<<grid1(S_ * DM_), 256, 0, stream>>>(x, xb, S_ * DM_);
  k_tconv<<<grid1(DM_ * DM_), 256, 0, stream>>>(Wq, WqT, DM_ * DM_, DM_, DM_, 0, DM_, 1);
  k_tconv<<<grid1(DM_ * DM_), 256, 0, stream>>>(Wk, WkT, DM_ * DM_, DM_, DM_, 0, DM_, 1);
  k_tconv<<<grid1(DM_ * DM_), 256, 0, stream>>>(Wv, WvT, DM_ * DM_, DM_, DM_, 0, DM_, 1);
  k_tconv<<<grid1(DM_ * DM_), 256, 0, stream>>>(Wo, WoT, DM_ * DM_, DM_, DM_, 0, DM_, 1);
  k_tconv<<<grid1(BS_ * HD_ * 4 * HD_), 256, 0, stream>>>(cW1, cW1T, BS_ * HD_ * 4 * HD_,
                                                          BS_ * HD_, 4 * HD_, 0, 4 * HD_, 1);
  k_tconv<<<grid1(4 * HD_ * HD_), 256, 0, stream>>>(cW2, cW2T, 4 * HD_ * HD_,
                                                    4 * HD_, HD_, 0, HD_, 1);
  k_tconv<<<grid1(DM_ * (DM_ / 2)), 256, 0, stream>>>(gW1, gW1T, DM_ * (DM_ / 2),
                                                      DM_, DM_ / 2, 0, DM_ / 2, 1);

  // ---- q/k/v projections ----
  gemm(xb, 0, DM_, WqT, 0, DM_, bq, qb, 0, DM_, S_, DM_, DM_, 1.f, 0, 1, 1);
  gemm(xb, 0, DM_, WkT, 0, DM_, bk, kf, 0, DM_, S_, DM_, DM_, 1.f, 0, 0, 1);
  gemm(xb, 0, DM_, WvT, 0, DM_, bv, vf, 0, DM_, S_, DM_, DM_, 1.f, 0, 0, 1);
  k_cvt<<<grid1(S_ * DM_), 256, 0, stream>>>(kf, kb, S_ * DM_);
  // vtb[h][d][s] = v[s][h*HD+d]
  k_tconv<<<grid1(H_ * HD_ * S_), 256, 0, stream>>>(vf, vtb, H_ * HD_ * S_,
                                                    S_, HD_, HD_, DM_, 1);

  // ---- branch 1: block compression (k then v through shared MLP buffers) ----
  k_buildKB<<<grid1(H_ * NB_ * BS_ * HD_), 256, 0, stream>>>(kf, pos, KB, H_ * NB_ * BS_ * HD_);
  gemm(KB, 0, BS_ * HD_, cW1T, 0, BS_ * HD_, cb1, H1, 0, 4 * HD_,
       H_ * NB_, 4 * HD_, BS_ * HD_, 1.f, 1, 0, 1);
  k_cvt<<<grid1(H_ * NB_ * 4 * HD_), 256, 0, stream>>>(H1, H1b, H_ * NB_ * 4 * HD_);
  gemm(H1b, 0, 4 * HD_, cW2T, 0, 4 * HD_, cb2, kcf, 0, HD_,
       H_ * NB_, HD_, 4 * HD_, 1.f, 0, 0, 1);
  k_cvt<<<grid1(H_ * NB_ * HD_), 256, 0, stream>>>(kcf, kcb, H_ * NB_ * HD_);

  k_buildKB<<<grid1(H_ * NB_ * BS_ * HD_), 256, 0, stream>>>(vf, pos, KB, H_ * NB_ * BS_ * HD_);
  gemm(KB, 0, BS_ * HD_, cW1T, 0, BS_ * HD_, cb1, H1, 0, 4 * HD_,
       H_ * NB_, 4 * HD_, BS_ * HD_, 1.f, 1, 0, 1);
  k_cvt<<<grid1(H_ * NB_ * 4 * HD_), 256, 0, stream>>>(H1, H1b, H_ * NB_ * 4 * HD_);
  gemm(H1b, 0, 4 * HD_, cW2T, 0, 4 * HD_, cb2, vcf, 0, HD_,
       H_ * NB_, HD_, 4 * HD_, 1.f, 0, 0, 1);
  // vctb[h][d][nb] = vc[h][nb][d]
  k_tconv<<<grid1(H_ * HD_ * NB_), 256, 0, stream>>>(vcf, vctb, H_ * HD_ * NB_,
                                                     NB_, HD_, (long)NB_ * HD_, HD_, 1);

  // ---- branch 2 selection: block means -> block scores -> top-k mask ----
  k_kbm<<<grid1(H_ * NB_ * HD_), 256, 0, stream>>>(kf, kbmb);
  gemm(qb, HD_, DM_, kbmb, (long)NB_ * HD_, HD_, nullptr, bsc, (long)S_ * NB_, NB_,
       S_, NB_, HD_, SCALE_, 0, 0, H_);
  k_topk<<<grid1(H_ * S_), 256, 0, stream>>>(bsc, msk, H_ * S_);

  // ---- three attention branches (flash, one wave per 16-query tile) ----
  dim3 fgrid(S_ / 16 / 4, H_);
  flash_wmma<<<fgrid, 128, 0, stream>>>(qb, DM_, kcb, (long)NB_ * HD_, HD_,
                                        vctb, (long)HD_ * NB_, NB_,
                                        outc, DM_, nullptr, NB_, S_, 2, SCALE_);
  flash_wmma<<<fgrid, 128, 0, stream>>>(qb, DM_, kb, (long)HD_, DM_,
                                        vtb, (long)HD_ * S_, S_,
                                        outs, DM_, msk, S_, S_, 0, SCALE_);
  flash_wmma<<<fgrid, 128, 0, stream>>>(qb, DM_, kb, (long)HD_, DM_,
                                        vtb, (long)HD_ * S_, S_,
                                        outw, DM_, nullptr, S_, S_, 1, SCALE_);

  // ---- gating + fusion + output projection ----
  gemm(xb, 0, DM_, gW1T, 0, DM_, gb1, h1g, 0, DM_ / 2, S_, DM_ / 2, DM_, 1.f, 1, 0, 1);
  k_gate<<<S_, 32, 0, stream>>>(h1g, gW2, gb2, gg);
  k_fuse<<<grid1(S_ * DM_), 256, 0, stream>>>(outc, outs, outw, gg, fusedb, S_ * DM_);
  gemm(fusedb, 0, DM_, WoT, 0, DM_, bo, (float*)d_out, 0, DM_, S_, DM_, DM_, 1.f, 0, 0, 1);
}
